// MultiHeadedAttention_28329604285102
// MI455X (gfx1250) — matrix operand order repaired and checked by reading, not compiled, not run on hardware
//
#include <hip/hip_runtime.h>
#include <hip/hip_bf16.h>
#include <stdint.h>

// Problem constants (fixed by the reference).
#define B_  4
#define S_  2048
#define D_  1024
#define H_  16
#define HD_ 64

typedef __attribute__((ext_vector_type(16))) __bf16 v16bf;
typedef __attribute__((ext_vector_type(8)))  __bf16 v8bf;
typedef __attribute__((ext_vector_type(8)))  float  v8f;
typedef __attribute__((ext_vector_type(4)))  unsigned int v4u;
typedef __attribute__((ext_vector_type(8)))  int v8i;
typedef __attribute__((ext_vector_type(4)))  int v4i;

#if defined(__has_builtin)
#  if __has_builtin(__builtin_amdgcn_tensor_load_to_lds)
#    define HAVE_TDM 1
#  endif
#endif

// ---------------------------------------------------------------- helpers --

__device__ __forceinline__ uint32_t lds_off_of(const void* p) {
  // low 32 bits of a flat LDS-aperture address == workgroup-relative LDS byte offset
  return (uint32_t)(uintptr_t)p;
}

// Build a v16bf A/B fragment from two 16-byte chunks.
__device__ __forceinline__ v16bf frag16(const void* p0, const void* p1) {
  union { v16bf v; v8bf h[2]; } u;
  u.h[0] = *(const v8bf*)p0;
  u.h[1] = *(const v8bf*)p1;
  return u.v;
}

__device__ __forceinline__ v8f wmma_bf16(v16bf a, v16bf b, v8f c) {
  return __builtin_amdgcn_wmma_f32_16x16x32_bf16(false, a, false, b, (short)0, c,
                                                 false, false);
}

__device__ __forceinline__ float redmax16(float v) {
  for (int m = 1; m <= 8; m <<= 1) v = fmaxf(v, __shfl_xor(v, m, 32));
  return v;
}
__device__ __forceinline__ float redsum16(float v) {
  for (int m = 1; m <= 8; m <<= 1) v += __shfl_xor(v, m, 32);
  return v;
}

// ------------------------------------------------- TDM 2D tile -> LDS stage --
// Loads `rows` rows of `rowElems` bf16 (global row stride `strideElems` elems)
// into LDS with a per-row pad of 8 bytes (pitch = rowElems*2 + 8).
// padIv: pad_interval code, 2^(padIv+1) DWORDs between pads (3 -> 64B, 4 -> 128B).

#ifdef HAVE_TDM
__device__ __forceinline__ void tdm_load_2d(uint32_t lds_off, const void* g,
                                            uint32_t rows, uint32_t rowElems,
                                            uint64_t strideElems, uint32_t padIv) {
  uint64_t ga = (uint64_t)(uintptr_t)g;
  v4u g0 = { 1u,                                   // count=1, user mode
             lds_off,
             (uint32_t)ga,
             (uint32_t)((ga >> 32) & 0x01FFFFFFu) | (2u << 30) };  // type=2
  uint32_t td0 = (uint32_t)strideElems;            // tensor_dim0 (row length)
  uint32_t td1 = 1u << 24;                         // generous: tiles always in-bounds
  v8i g1 = {
    (int)((1u << 16) |                             // data_size = 2 bytes
          (1u << 20) |                             // pad_enable
          (padIv << 22) |                          // pad_interval
          (1u << 25)),                             // pad_amount = 2 DWORDs (8B)
    (int)((td0 & 0xFFFFu) << 16),                  // abar_addr=0 | tensor_dim0.lo
    (int)((td0 >> 16) | ((td1 & 0xFFFFu) << 16)),  // tensor_dim0.hi | tensor_dim1.lo
    (int)((td1 >> 16) | (rowElems << 16)),         // tensor_dim1.hi | tile_dim0
    (int)(rows & 0xFFFFu),                         // tile_dim1 | tile_dim2=0
    (int)(uint32_t)(strideElems & 0xFFFFFFFFu),    // tensor_dim0_stride.lo
    (int)(uint32_t)((strideElems >> 32) & 0xFFFFu),
    0 };
  v4i z4 = { 0, 0, 0, 0 };
  v8i z8 = { 0, 0, 0, 0, 0, 0, 0, 0 };
  // clang-23 / therock-10.0 arity: (g0, g1, g2, g3, g4, cpol)
  __builtin_amdgcn_tensor_load_to_lds(g0, g1, z4, z4, z8, 0);
}
#endif

__device__ __forceinline__ void stage_tile(char* lds, const __bf16* g,
                                           uint32_t rows, uint32_t rowElems,
                                           uint32_t strideElems, uint32_t padIv) {
#ifdef HAVE_TDM
  if (threadIdx.x == 0)
    tdm_load_2d(lds_off_of(lds), g, rows, rowElems, strideElems, padIv);
#else
  const uint32_t pitch  = rowElems * 2 + 8;
  const uint32_t chunks = rowElems / 8;            // 16B chunks per row
  for (uint32_t i = threadIdx.x; i < rows * chunks; i += blockDim.x) {
    uint32_t r = i / chunks, c = i - r * chunks;
    *(uint4*)(lds + r * pitch + c * 16) =
        *(const uint4*)(g + (size_t)r * strideElems + c * 8);
  }
#endif
}

__device__ __forceinline__ void stage_commit() {
#ifdef HAVE_TDM
  if (threadIdx.x == 0) {
#  if __has_builtin(__builtin_amdgcn_s_wait_tensorcnt)
    __builtin_amdgcn_s_wait_tensorcnt(0);
#  else
    asm volatile("s_wait_tensorcnt 0" ::: "memory");
#  endif
  }
#endif
  __syncthreads();
}

// ----------------------------------------------------------- f32 -> bf16 ----

__global__ void cvt_bf16(const float* __restrict__ src, __bf16* __restrict__ dst,
                         int n) {
  int i = (blockIdx.x * blockDim.x + threadIdx.x) * 4;
  if (i + 3 < n) {
    float4 f = *(const float4*)(src + i);
    dst[i + 0] = (__bf16)f.x;
    dst[i + 1] = (__bf16)f.y;
    dst[i + 2] = (__bf16)f.z;
    dst[i + 3] = (__bf16)f.w;
  }
}

// ------------------------------------------------------------- WMMA GEMM ----
// Y = X(bf16, MxK row-major) @ W(bf16, NxK row-major)^T
// MODE 0: store bf16 as Q/K layout [b,h,s,hd]
// MODE 1: store bf16 as V-transposed layout [b,h,hd,s]
// MODE 2: store f32 YF[m*D+n] = acc + bias[n]
// Tiles: 128x128 per WG (8 waves, wave = 2 Mtiles x 4 Ntiles), BK=32.

#define GT 9216       // one staged tile: 128 rows * (32*2+8) bytes
template <int MODE>
__global__ __launch_bounds__(256) void gemm_wmma(
    const __bf16* __restrict__ X, const __bf16* __restrict__ W,
    __bf16* __restrict__ YB, float* __restrict__ YF,
    const float* __restrict__ bias) {
  __shared__ char smem[4 * GT];                    // [buf][X|W]
  const uint32_t tid  = threadIdx.x;
  const uint32_t lane = tid & 31, wave = tid >> 5;
  const uint32_t half = lane >> 4, l16 = lane & 15;
  const uint32_t mBase = blockIdx.y * 128 + (wave & 3) * 32;
  const uint32_t nBase = blockIdx.x * 128 + (wave >> 2) * 64;

  v8f acc[2][4] = {};
  const uint32_t KIT = D_ / 32;

  const __bf16* Xb = X + (size_t)(blockIdx.y * 128) * D_;
  const __bf16* Wb = W + (size_t)(blockIdx.x * 128) * D_;

  stage_tile(smem + 0,  Xb, 128, 32, D_, 3);
  stage_tile(smem + GT, Wb, 128, 32, D_, 3);

  for (uint32_t kk = 0; kk < KIT; ++kk) {
    stage_commit();                                // data for kk is ready
    const uint32_t cur = kk & 1, nxt = cur ^ 1;
    if (kk + 1 < KIT) {                            // prefetch kk+1 into other buf
      stage_tile(smem + nxt * 2 * GT,      Xb + (kk + 1) * 32, 128, 32, D_, 3);
      stage_tile(smem + nxt * 2 * GT + GT, Wb + (kk + 1) * 32, 128, 32, D_, 3);
    }
    const char* lx = smem + cur * 2 * GT;
    const char* lw = smem + cur * 2 * GT + GT;

    v16bf bfr[4];
    for (int nt = 0; nt < 4; ++nt) {
      const char* rp = lw + ((wave >> 2) * 64 + nt * 16 + l16) * 72 + half * 32;
      bfr[nt] = frag16(rp, rp + 16);               // contiguous 16 K elems
    }
    for (int mt = 0; mt < 2; ++mt) {
      const char* rp = lx + ((wave & 3) * 32 + mt * 16 + l16) * 72 + half * 32;
      v16bf a = frag16(rp, rp + 16);               // contiguous 16 K elems
      for (int nt = 0; nt < 4; ++nt)
        acc[mt][nt] = wmma_bf16(a, bfr[nt], acc[mt][nt]);
    }
  }

  for (int mt = 0; mt < 2; ++mt)
    for (int nt = 0; nt < 4; ++nt)
      for (int r = 0; r < 8; ++r) {
        uint32_t m = mBase + mt * 16 + r + half * 8;
        uint32_t n = nBase + nt * 16 + l16;
        float v = acc[mt][nt][r];
        if (MODE == 2) {
          YF[(size_t)m * D_ + n] = v + bias[n];
        } else {
          uint32_t b = m >> 11, s = m & (S_ - 1), h = n >> 6, d = n & 63;
          if (MODE == 0)
            YB[((((size_t)b * H_ + h) * S_ + s) << 6) + d] = (__bf16)v;
          else
            YB[(((size_t)b * H_ + h) * HD_ + d) * S_ + s] = (__bf16)v;
        }
      }
}

// -------------------------------------------------------- flash attention ----
// Grid: (S/128, H, B); 8 waves, each wave owns 16 query rows.
// K/V blocks of 64 keys, TDM double-buffered. Vt layout [b,h,hd,s].

#define KVT 8704      // 64 rows * (64*2+8) bytes
__global__ __launch_bounds__(256) void flash_attn(
    const __bf16* __restrict__ Q, const __bf16* __restrict__ K,
    const __bf16* __restrict__ Vt, __bf16* __restrict__ ctx) {
  __shared__ char smem[4 * KVT + 8 * 16 * 136];    // K0 V0 K1 V1 | P (per-wave)
  const uint32_t tid = threadIdx.x, lane = tid & 31, wave = tid >> 5;
  const uint32_t half = lane >> 4, l16 = lane & 15;
  const uint32_t qb = blockIdx.x, h = blockIdx.y, b = blockIdx.z;
  const uint32_t rowBase = qb * 128 + wave * 16;

  const __bf16* Qh = Q  + (((size_t)b * H_ + h) * S_) * HD_;
  const __bf16* Kh = K  + (((size_t)b * H_ + h) * S_) * HD_;
  const __bf16* Vh = Vt + (((size_t)b * H_ + h) * HD_) * S_;

  // Q fragments held in registers for the whole kernel (row = rowBase + l16).
  const char* qrow = (const char*)(Qh + (size_t)(rowBase + l16) * HD_);
  v16bf qa[2];
  for (int ks = 0; ks < 2; ++ks) {
    const char* p = qrow + ks * 64 + half * 32;
    qa[ks] = frag16(p, p + 16);
  }

  v8f o[4] = {};
  float mrow[8], lrow[8];
  for (int r = 0; r < 8; ++r) { mrow[r] = -3.0e38f; lrow[r] = 0.f; }

  const uint32_t jcount = qb * 2 + 2;              // causal: keys <= qb*128+127

  stage_tile(smem + 0,   Kh, 64, 64, HD_, 4);
  stage_tile(smem + KVT, Vh, 64, 64, S_, 4);

  for (uint32_t j = 0; j < jcount; ++j) {
    stage_commit();
    const uint32_t cur = j & 1, nxt = cur ^ 1;
    if (j + 1 < jcount) {
      stage_tile(smem + nxt * 2 * KVT,       Kh + (size_t)(j + 1) * 64 * HD_,
                 64, 64, HD_, 4);
      stage_tile(smem + nxt * 2 * KVT + KVT, Vh + (j + 1) * 64, 64, 64, S_, 4);
    }
    const char* kb = smem + cur * 2 * KVT;
    const char* vb = smem + cur * 2 * KVT + KVT;

    // S = Q K^T  (4 tiles of 16 keys, 2 k-steps over hd)
    v8f s4[4];
    for (int t = 0; t < 4; ++t) {
      v8f c = {};
      for (int ks = 0; ks < 2; ++ks) {
        const char* rp = kb + (t * 16 + l16) * 136 + ks * 64 + half * 32;
        c = wmma_bf16(qa[ks], frag16(rp, rp + 16), c);
      }
      s4[t] = c;
    }

    const uint32_t jbase = j * 64;
    const bool needMask = (jbase + 63) > (rowBase + 15);
    float pv[4][8];
    for (int r = 0; r < 8; ++r) {
      const uint32_t qg = rowBase + r + half * 8;
      float mx = -3.0e38f;
      for (int t = 0; t < 4; ++t) {
        float v = s4[t][r] * 0.125f;               // 1/sqrt(64)
        if (needMask && (jbase + t * 16 + l16) > qg) v = -3.0e38f;
        pv[t][r] = v;
        mx = fmaxf(mx, v);
      }
      mx = redmax16(mx);
      const float mnew  = fmaxf(mrow[r], mx);
      const float alpha = __expf(mrow[r] - mnew);
      mrow[r] = mnew;
      float rs = 0.f;
      for (int t = 0; t < 4; ++t) {
        float e = __expf(pv[t][r] - mnew);
        pv[t][r] = e;
        rs += e;
      }
      rs = redsum16(rs);
      lrow[r] = lrow[r] * alpha + rs;
      for (int t2 = 0; t2 < 4; ++t2) o[t2][r] *= alpha;
    }

    // C-fragment -> A-fragment relayout of P through per-wave LDS scratch.
    char* Pw = smem + 4 * KVT + wave * (16 * 136);
    for (int r = 0; r < 8; ++r)
      for (int t = 0; t < 4; ++t)
        *(__bf16*)(Pw + (r + half * 8) * 136 + (t * 16 + l16) * 2) =
            (__bf16)pv[t][r];
    asm volatile("s_wait_dscnt 0" ::: "memory");   // wave-local LDS RAW

    // O += P V  (contraction over the 64 keys, 2 k-steps)
    for (int ks = 0; ks < 2; ++ks) {
      const char* rp = Pw + l16 * 136 + ks * 64 + half * 32;
      v16bf pa = frag16(rp, rp + 16);
      for (int t2 = 0; t2 < 4; ++t2) {
        const char* vp = vb + (t2 * 16 + l16) * 136 + ks * 64 + half * 32;
        o[t2] = wmma_bf16(pa, frag16(vp, vp + 16), o[t2]);
      }
    }
  }

  // ctx[b, s, h*64+d] in bf16 for the output projection GEMM.
  for (int t2 = 0; t2 < 4; ++t2)
    for (int r = 0; r < 8; ++r) {
      const uint32_t srow = rowBase + r + half * 8;
      ctx[((size_t)b * S_ + srow) * D_ + h * HD_ + t2 * 16 + l16] =
          (__bf16)(o[t2][r] / lrow[r]);
    }
}

// ---------------------------------------------------------------- launcher --

extern "C" void kernel_launch(void* const* d_in, const int* in_sizes, int n_in,
                              void* d_out, int out_size, void* d_ws, size_t ws_size,
                              hipStream_t stream) {
  const float* x  = (const float*)d_in[0];
  const float* Wq = (const float*)d_in[1];
  const float* Wk = (const float*)d_in[2];
  const float* Wv = (const float*)d_in[3];
  const float* Wo = (const float*)d_in[4];
  const float* bo = (const float*)d_in[5];
  float* out = (float*)d_out;

  char* ws = (char*)d_ws;
  const size_t NX = (size_t)B_ * S_ * D_;          // 8Mi elems -> 16MB bf16
  const size_t NW = (size_t)D_ * D_;               // 1Mi elems -> 2MB bf16
  __bf16* xbf  = (__bf16*)(ws + 0 * NX);
  __bf16* qbf  = (__bf16*)(ws + 2 * NX);
  __bf16* kbf  = (__bf16*)(ws + 4 * NX);
  __bf16* vtbf = (__bf16*)(ws + 6 * NX);
  __bf16* cbf  = (__bf16*)(ws + 8 * NX);
  __bf16* wqb  = (__bf16*)(ws + 10 * NX + 0 * NW);
  __bf16* wkb  = (__bf16*)(ws + 10 * NX + 2 * NW);
  __bf16* wvb  = (__bf16*)(ws + 10 * NX + 4 * NW);
  __bf16* wob  = (__bf16*)(ws + 10 * NX + 6 * NW); // total ~88MB of d_ws

  const int T = 256;
  cvt_bf16<<<(int)(NX / 4 / T), T, 0, stream>>>(x,  xbf, (int)NX);
  cvt_bf16<<<(int)(NW / 4 / T), T, 0, stream>>>(Wq, wqb, (int)NW);
  cvt_bf16<<<(int)(NW / 4 / T), T, 0, stream>>>(Wk, wkb, (int)NW);
  cvt_bf16<<<(int)(NW / 4 / T), T, 0, stream>>>(Wv, wvb, (int)NW);
  cvt_bf16<<<(int)(NW / 4 / T), T, 0, stream>>>(Wo, wob, (int)NW);

  dim3 gg(D_ / 128, (B_ * S_) / 128);              // (8, 64)
  gemm_wmma<0><<<gg, 256, 0, stream>>>(xbf, wqb, qbf,  nullptr, nullptr);
  gemm_wmma<0><<<gg, 256, 0, stream>>>(xbf, wkb, kbf,  nullptr, nullptr);
  gemm_wmma<1><<<gg, 256, 0, stream>>>(xbf, wvb, vtbf, nullptr, nullptr);

  flash_attn<<<dim3(S_ / 128, H_, B_), 256, 0, stream>>>(qbf, kbf, vtbf, cbf);

  gemm_wmma<2><<<gg, 256, 0, stream>>>(cbf, wob, nullptr, out, bo);
}
